// CNV_70626442216003
// MI455X (gfx1250) — 8-bit matrix operand order repaired and checked by reading, compiled, not run on hardware
//
#include <hip/hip_runtime.h>
#include <stdint.h>

#define EPSV 1e-4f
#define BATCH 2048

typedef int v8i __attribute__((ext_vector_type(8)));

__device__ __forceinline__ int8_t bsign(float x) { return x >= 0.f ? (int8_t)1 : (int8_t)-1; }

__global__ void qconv0(const float* __restrict__ x, const float* __restrict__ w,
                       const float* __restrict__ g, const float* __restrict__ be,
                       const float* __restrict__ mu, const float* __restrict__ va,
                       int8_t* __restrict__ out)
{
    const int HO = 62, WO = 62, HI = 64, WI = 64, CI = 3, CO = 4;
    long idx = (long)blockIdx.x * blockDim.x + threadIdx.x;
    long total = (long)BATCH * CO * HO * WO;
    if (idx >= total) return;
    int xw = (int)(idx % WO); long t = idx / WO;
    int yh = (int)(t % HO); t /= HO;
    int co = (int)(t % CO); long b = t / CO;
    int acc = 0;
    for (int ci = 0; ci < CI; ++ci)
        for (int kh = 0; kh < 3; ++kh)
            for (int kw = 0; kw < 3; ++kw) {
                float xv = x[((b * CI + ci) * HI + yh + kh) * WI + xw + kw];
                float q = rintf((2.f * xv - 1.f) * 128.f);
                q = fminf(fmaxf(q, -128.f), 127.f);
                int ws = (w[((co * CI + ci) * 3 + kh) * 3 + kw] >= 0.f) ? 1 : -1;
                acc += (int)q * ws;
            }
    float xf = (float)acc * 0.0078125f;
    float s = g[co] * rsqrtf(va[co] + EPSV);
    float y = (xf - mu[co]) * s + be[co];
    out[((b * CO + co) * HO + yh) * WO + xw] = bsign(y);
}

template <int CI, int CO, int HI, int WI>
__global__ void bconv(const int8_t* __restrict__ in, const float* __restrict__ w,
                      const float* __restrict__ g, const float* __restrict__ be,
                      const float* __restrict__ mu, const float* __restrict__ va,
                      int8_t* __restrict__ out, int outStride)
{
    const int HO = HI - 2, WO = WI - 2;
    long idx = (long)blockIdx.x * blockDim.x + threadIdx.x;
    long total = (long)BATCH * CO * HO * WO;
    if (idx >= total) return;
    int xw = (int)(idx % WO); long t = idx / WO;
    int yh = (int)(t % HO); t /= HO;
    int co = (int)(t % CO); long b = t / CO;
    int acc = 0;
    for (int ci = 0; ci < CI; ++ci)
        for (int kh = 0; kh < 3; ++kh)
            for (int kw = 0; kw < 3; ++kw) {
                int av = (int)in[((b * CI + ci) * HI + yh + kh) * WI + xw + kw];
                int ws = (w[((co * CI + ci) * 3 + kh) * 3 + kw] >= 0.f) ? 1 : -1;
                acc += av * ws;
            }
    float s = g[co] * rsqrtf(va[co] + EPSV);
    float y = ((float)acc - mu[co]) * s + be[co];
    out[b * (long)outStride + (co * HO + yh) * WO + xw] = bsign(y);
}

template <int C, int HI, int WI>
__global__ void maxpool2(const int8_t* __restrict__ in, int8_t* __restrict__ out)
{
    const int HO = HI / 2, WO = WI / 2;
    long idx = (long)blockIdx.x * blockDim.x + threadIdx.x;
    long total = (long)BATCH * C * HO * WO;
    if (idx >= total) return;
    int xw = (int)(idx % WO); long t = idx / WO;
    int yh = (int)(t % HO); t /= HO;
    int c = (int)(t % C); long b = t / C;
    long base = ((b * C + c) * HI + 2 * yh) * WI + 2 * xw;
    int8_t m = in[base];
    m = max(m, in[base + 1]);
    m = max(m, in[base + WI]);
    m = max(m, in[base + WI + 1]);
    out[((b * C + c) * HO + yh) * WO + xw] = m;
}

__global__ void packw(const float* __restrict__ wf, int8_t* __restrict__ out,
                      int N, int K, int Npad, int Kpad)
{
    long idx = (long)blockIdx.x * blockDim.x + threadIdx.x;
    long total = (long)Npad * Kpad;
    if (idx >= total) return;
    int k = (int)(idx % Kpad);
    int n = (int)(idx / Kpad);
    int8_t v = 0;
    if (n < N && k < K) v = (wf[(long)n * K + k] >= 0.f) ? (int8_t)1 : (int8_t)-1;
    out[idx] = v;
}

__global__ void __launch_bounds__(128)
bgemm_bnsign(const int8_t* __restrict__ A, const int8_t* __restrict__ Bw,
             int Kpad, int Npad, int Nact,
             const float* __restrict__ g, const float* __restrict__ be,
             const float* __restrict__ mu, const float* __restrict__ va,
             int8_t* __restrict__ out, int outStride)
{
    int wave = threadIdx.x >> 5;
    int lane = threadIdx.x & 31;
    int nTiles = Npad >> 4;
    int tile = blockIdx.x * 4 + wave;
    if (tile >= (BATCH / 16) * nTiles) return;
    int mTile = tile / nTiles, nTile = tile % nTiles;
    int lm = lane & 15, half = lane >> 4;
    const int8_t* aRow = A + (size_t)(mTile * 16 + lm) * Kpad;
    const int8_t* bRow = Bw + (size_t)(nTile * 16 + lm) * Kpad;
    v8i acc = {};
    for (int kk = 0; kk < Kpad; kk += 64) {
        v8i av, bv;
#pragma unroll
        for (int v = 0; v < 8; ++v) {
            int ao = ((v >> 1) << 4) + (half << 3) + ((v & 1) << 2);
            int bo = ((v >> 1) << 4) + (half << 3) + ((v & 1) << 2);
            av[v] = *reinterpret_cast<const int*>(aRow + kk + ao);
            bv[v] = *reinterpret_cast<const int*>(bRow + kk + bo);
        }
        acc = __builtin_amdgcn_wmma_i32_16x16x64_iu8(true, av, true, bv, acc, false, false);
    }
    int N = nTile * 16 + lm;
    if (N >= Nact) return;
    float s = g[N] * rsqrtf(va[N] + EPSV);
    float muv = mu[N], bev = be[N];
#pragma unroll
    for (int r = 0; r < 8; ++r) {
        int M = mTile * 16 + r + (half << 3);
        float y = ((float)acc[r] - muv) * s + bev;
        out[(size_t)M * outStride + N] = bsign(y);
    }
}

__global__ void __launch_bounds__(128)
bgemm_tn(const int8_t* __restrict__ A, const int8_t* __restrict__ Bw,
         int Kpad, int Npad, int Nact,
         const float* __restrict__ tw, const float* __restrict__ tb,
         const float* __restrict__ tm, const float* __restrict__ tv,
         float* __restrict__ out, int outStride)
{
    int wave = threadIdx.x >> 5;
    int lane = threadIdx.x & 31;
    int nTiles = Npad >> 4;
    int tile = blockIdx.x * 4 + wave;
    if (tile >= (BATCH / 16) * nTiles) return;
    int mTile = tile / nTiles, nTile = tile % nTiles;
    int lm = lane & 15, half = lane >> 4;
    const int8_t* aRow = A + (size_t)(mTile * 16 + lm) * Kpad;
    const int8_t* bRow = Bw + (size_t)(nTile * 16 + lm) * Kpad;
    v8i acc = {};
    for (int kk = 0; kk < Kpad; kk += 64) {
        v8i av, bv;
#pragma unroll
        for (int v = 0; v < 8; ++v) {
            int ao = ((v >> 1) << 4) + (half << 3) + ((v & 1) << 2);
            int bo = ((v >> 1) << 4) + (half << 3) + ((v & 1) << 2);
            av[v] = *reinterpret_cast<const int*>(aRow + kk + ao);
            bv[v] = *reinterpret_cast<const int*>(bRow + kk + bo);
        }
        acc = __builtin_amdgcn_wmma_i32_16x16x64_iu8(true, av, true, bv, acc, false, false);
    }
    int N = nTile * 16 + lm;
    if (N >= Nact) return;
    float scale = tw[0] * rsqrtf(tv[0] + EPSV);
    float tmv = tm[0], tbv = tb[0];
#pragma unroll
    for (int r = 0; r < 8; ++r) {
        int M = mTile * 16 + r + (half << 3);
        out[(size_t)M * outStride + N] = ((float)acc[r] - tmv) * scale + tbv;
    }
}

static inline int blks(long n, int t) { return (int)((n + t - 1) / t); }

extern "C" void kernel_launch(void* const* d_in, const int* in_sizes, int n_in,
                              void* d_out, int out_size, void* d_ws, size_t ws_size,
                              hipStream_t stream)
{
    (void)in_sizes; (void)n_in; (void)out_size; (void)ws_size;
    const float* x = (const float*)d_in[0];
    const float* convw[6]; for (int i = 0; i < 6; ++i) convw[i] = (const float*)d_in[1 + i];
    const float* cbn[6][4];
    for (int i = 0; i < 6; ++i) for (int j = 0; j < 4; ++j) cbn[i][j] = (const float*)d_in[7 + 4 * i + j];
    const float* fcw[3]; for (int i = 0; i < 3; ++i) fcw[i] = (const float*)d_in[31 + i];
    const float* fbn[2][4];
    for (int i = 0; i < 2; ++i) for (int j = 0; j < 4; ++j) fbn[i][j] = (const float*)d_in[34 + 4 * i + j];
    const float* tw = (const float*)d_in[42];
    const float* tb = (const float*)d_in[43];
    const float* tm = (const float*)d_in[44];
    const float* tv = (const float*)d_in[45];
    float* out = (float*)d_out;

    uint8_t* ws = (uint8_t*)d_ws;
    const size_t OFF_PING = 0;
    const size_t OFF_PONG = 33554432;
    const size_t OFF_FCIN = 67108864;
    const size_t OFF_FC1  = OFF_FCIN + (size_t)2048 * 1344;
    const size_t OFF_FC2  = OFF_FC1  + (size_t)2048 * 768;
    const size_t OFF_W1   = OFF_FC2  + (size_t)2048 * 512;
    const size_t OFF_W2   = OFF_W1   + (size_t)752 * 1344;
    const size_t OFF_W3   = OFF_W2   + (size_t)512 * 768;
    int8_t* ping = (int8_t*)(ws + OFF_PING);
    int8_t* pong = (int8_t*)(ws + OFF_PONG);
    int8_t* fcin = (int8_t*)(ws + OFF_FCIN);
    int8_t* fc1  = (int8_t*)(ws + OFF_FC1);
    int8_t* fc2  = (int8_t*)(ws + OFF_FC2);
    int8_t* w1b  = (int8_t*)(ws + OFF_W1);
    int8_t* w2b  = (int8_t*)(ws + OFF_W2);
    int8_t* w3b  = (int8_t*)(ws + OFF_W3);

    const int T = 256;

    hipMemsetAsync(fcin, 0, (size_t)2048 * 1344, stream);
    hipMemsetAsync(fc1,  0, (size_t)2048 * 768,  stream);
    hipMemsetAsync(fc2,  0, (size_t)2048 * 512,  stream);

    packw<<<blks((long)752 * 1344, T), T, 0, stream>>>(fcw[0], w1b, 750, 1296, 752, 1344);
    packw<<<blks((long)512 * 768,  T), T, 0, stream>>>(fcw[1], w2b, 500, 750,  512, 768);
    packw<<<blks((long)208 * 512,  T), T, 0, stream>>>(fcw[2], w3b, 200, 500,  208, 512);

    qconv0<<<blks((long)BATCH * 4 * 62 * 62, T), T, 0, stream>>>(
        x, convw[0], cbn[0][0], cbn[0][1], cbn[0][2], cbn[0][3], ping);
    bconv<4, 4, 62, 62><<<blks((long)BATCH * 4 * 60 * 60, T), T, 0, stream>>>(
        ping, convw[1], cbn[1][0], cbn[1][1], cbn[1][2], cbn[1][3], pong, 4 * 60 * 60);
    maxpool2<4, 60, 60><<<blks((long)BATCH * 4 * 30 * 30, T), T, 0, stream>>>(pong, ping);
    bconv<4, 8, 30, 30><<<blks((long)BATCH * 8 * 28 * 28, T), T, 0, stream>>>(
        ping, convw[2], cbn[2][0], cbn[2][1], cbn[2][2], cbn[2][3], pong, 8 * 28 * 28);
    bconv<8, 8, 28, 28><<<blks((long)BATCH * 8 * 26 * 26, T), T, 0, stream>>>(
        pong, convw[3], cbn[3][0], cbn[3][1], cbn[3][2], cbn[3][3], ping, 8 * 26 * 26);
    maxpool2<8, 26, 26><<<blks((long)BATCH * 8 * 13 * 13, T), T, 0, stream>>>(ping, pong);
    bconv<8, 16, 13, 13><<<blks((long)BATCH * 16 * 11 * 11, T), T, 0, stream>>>(
        pong, convw[4], cbn[4][0], cbn[4][1], cbn[4][2], cbn[4][3], ping, 16 * 11 * 11);
    bconv<16, 16, 11, 11><<<blks((long)BATCH * 16 * 9 * 9, T), T, 0, stream>>>(
        ping, convw[5], cbn[5][0], cbn[5][1], cbn[5][2], cbn[5][3], fcin, 1344);

    {
        int tiles = (BATCH / 16) * (752 / 16);
        bgemm_bnsign<<<(tiles + 3) / 4, 128, 0, stream>>>(
            fcin, w1b, 1344, 752, 750,
            fbn[0][0], fbn[0][1], fbn[0][2], fbn[0][3], fc1, 768);
    }
    {
        int tiles = (BATCH / 16) * (512 / 16);
        bgemm_bnsign<<<(tiles + 3) / 4, 128, 0, stream>>>(
            fc1, w2b, 768, 512, 500,
            fbn[1][0], fbn[1][1], fbn[1][2], fbn[1][3], fc2, 512);
    }
    {
        int tiles = (BATCH / 16) * (208 / 16);
        bgemm_tn<<<(tiles + 3) / 4, 128, 0, stream>>>(
            fc2, w3b, 512, 208, 200, tw, tb, tm, tv, out, 200);
    }
}
